// TestNN_26431228739616
// MI455X (gfx1250) — matrix operand order repaired and checked by reading, not compiled, not run on hardware
//
#include <hip/hip_runtime.h>

typedef __attribute__((ext_vector_type(16))) __bf16 v16bf;
typedef __attribute__((ext_vector_type(8)))  float  v8f;
typedef __attribute__((ext_vector_type(4)))  int    v4i;

#define BB 16
#define NN 1000
#define EE 4000
#define LL 64
#define AAC 128
#define DD 256
#define HD4 1024
#define QK 4096
#define SCALE_ 0.0625f   // 1/sqrt(256)

#if __has_builtin(__builtin_amdgcn_global_load_async_to_lds_b128) && \
    __has_builtin(__builtin_amdgcn_s_wait_asynccnt)
#define USE_ASYNC_LDS 1
#else
#define USE_ASYNC_LDS 0
#endif

#define GLOBAL_AS __attribute__((address_space(1)))
#define LDS_AS    __attribute__((address_space(3)))

__device__ __forceinline__ float sigm(float x) { return 1.0f / (1.0f + __expf(-x)); }

// ---------------------------------------------------------------------------
// Weight prep: f32 [K,N] -> bf16 [N,K]  (so WMMA B-fragments are k-contiguous)
// ---------------------------------------------------------------------------
__global__ void k_convert_t(const float* __restrict__ W, __bf16* __restrict__ Bt,
                            int K, int N) {
  int idx = blockIdx.x * 256 + threadIdx.x;
  if (idx >= K * N) return;
  int n = idx / K, k = idx - n * K;
  Bt[(size_t)n * K + k] = (__bf16)W[(size_t)k * N + n];
}

__global__ void k_convert(const float* __restrict__ src, __bf16* __restrict__ dst, int n) {
  int idx = blockIdx.x * 256 + threadIdx.x;
  if (idx < n) dst[idx] = (__bf16)src[idx];
}

__global__ void k_fill_int(int* __restrict__ p, int v, int n) {
  int idx = blockIdx.x * 256 + threadIdx.x;
  if (idx < n) p[idx] = v;
}

// ---------------------------------------------------------------------------
// Embedding gather -> bf16 rows
// ---------------------------------------------------------------------------
__global__ void k_gather(const int* __restrict__ tok, const float* __restrict__ emb,
                         __bf16* __restrict__ out, int rows) {
  int idx = blockIdx.x * 256 + threadIdx.x;
  if (idx >= rows * DD) return;
  int r = idx >> 8, d = idx & 255;
  out[idx] = (__bf16)emb[(size_t)tok[r] * DD + d];
}

// mixed_embed: emb[tok] + graph_repr[b, node]
__global__ void k_mixed(const int* __restrict__ tok, const int* __restrict__ node,
                        const float* __restrict__ emb, const float* __restrict__ graph,
                        __bf16* __restrict__ outbf, float* __restrict__ outf,
                        int rows, int rowsPerB) {
  int idx = blockIdx.x * 256 + threadIdx.x;
  if (idx >= rows * DD) return;
  int r = idx >> 8, d = idx & 255;
  int b = r / rowsPerB;
  float v = emb[(size_t)tok[r] * DD + d] + graph[((size_t)b * NN + node[r]) * DD + d];
  outbf[idx] = (__bf16)v;
  if (outf) outf[idx] = v;
}

// ---------------------------------------------------------------------------
// bf16 WMMA GEMM: C[M,N] = act(A[M,K] @ Bt[N,K]^T + bias)
// 8 wave32s, 64x64 block tile, K-step 32, double-buffered LDS.
// Staging uses GLOBAL_LOAD_ASYNC_TO_LDS_B128 (ASYNCcnt path, no VGPR
// round-trip); falls back to global_load_b128 + ds_store_b128 if the builtin
// is missing. One barrier per K-step either way.
// ---------------------------------------------------------------------------
__global__ __launch_bounds__(256) void k_gemm(
    const __bf16* __restrict__ A, const __bf16* __restrict__ Bt,
    const float* __restrict__ bias, float* __restrict__ C, __bf16* __restrict__ Cbf,
    int M, int N, int K, int act) {
  __shared__ __bf16 Alds[2][64][32];
  __shared__ __bf16 Blds[2][64][32];
  const int tid  = threadIdx.x;
  const int wave = tid >> 5;
  const int lane = tid & 31;
  const int m0 = blockIdx.y * 64;
  const int n0 = blockIdx.x * 64;
  const int nsub  = wave & 3;          // 4 N-subtiles
  const int msub0 = (wave >> 2) * 2;   // each wave owns 2 stacked M-subtiles
  v8f acc0 = {}; v8f acc1 = {};

  const int lrow  = tid >> 2;          // 64 rows, 4 threads/row
  const int lkoff = (tid & 3) * 8;     // 8 bf16 = 16B per thread
  const int lh    = (lane >> 4) & 1;
  const int l15   = lane & 15;

#if USE_ASYNC_LDS
#define STAGE(buf, kk)                                                          \
  do {                                                                          \
    const __bf16* ag_ = A  + (size_t)(m0 + lrow) * K + (kk) + lkoff;            \
    const __bf16* bg_ = Bt + (size_t)(n0 + lrow) * K + (kk) + lkoff;            \
    __builtin_prefetch(ag_ + 64, 0, 1);                                         \
    __builtin_prefetch(bg_ + 64, 0, 1);                                         \
    __builtin_amdgcn_global_load_async_to_lds_b128(                             \
        (GLOBAL_AS v4i*)(uintptr_t)ag_,                                         \
        (LDS_AS v4i*)(uintptr_t)&Alds[buf][lrow][lkoff], 0, 0);                 \
    __builtin_amdgcn_global_load_async_to_lds_b128(                             \
        (GLOBAL_AS v4i*)(uintptr_t)bg_,                                         \
        (LDS_AS v4i*)(uintptr_t)&Blds[buf][lrow][lkoff], 0, 0);                 \
  } while (0)
#define WAIT_STAGE() __builtin_amdgcn_s_wait_asynccnt(0)
#else
#define STAGE(buf, kk)                                                          \
  do {                                                                          \
    const __bf16* ag_ = A  + (size_t)(m0 + lrow) * K + (kk) + lkoff;            \
    const __bf16* bg_ = Bt + (size_t)(n0 + lrow) * K + (kk) + lkoff;            \
    __builtin_prefetch(ag_ + 64, 0, 1);                                         \
    __builtin_prefetch(bg_ + 64, 0, 1);                                         \
    *(uint4*)&Alds[buf][lrow][lkoff] = *(const uint4*)ag_;                      \
    *(uint4*)&Blds[buf][lrow][lkoff] = *(const uint4*)bg_;                      \
  } while (0)
#define WAIT_STAGE() ((void)0)
#endif

  STAGE(0, 0);
  WAIT_STAGE();
  __syncthreads();

  int cur = 0;
  for (int k0 = 0; k0 < K; k0 += 32) {
    if (k0 + 32 < K) STAGE(cur ^ 1, k0 + 32);   // overlap next tile with WMMAs

    // B fragment: column n fixed per lane, contiguous K halves
    v16bf bfrag = *(const v16bf*)&Blds[cur][nsub * 16 + l15][lh * 16];
    // A fragments, contiguous K halves to match B (lane 0-15: K 0-15, lane 16-31: K 16-31)
    v16bf afrag0, afrag1;
#pragma unroll
    for (int i = 0; i < 16; ++i) {
      int j = i >> 1, h = i & 1;
      int kk = lh * 16 + i; (void)j; (void)h;
      afrag0[i] = Alds[cur][msub0 * 16 + l15][kk];
      afrag1[i] = Alds[cur][msub0 * 16 + 16 + l15][kk];
    }
    acc0 = __builtin_amdgcn_wmma_f32_16x16x32_bf16(false, afrag0, false, bfrag,
                                                   (short)0, acc0, false, false);
    acc1 = __builtin_amdgcn_wmma_f32_16x16x32_bf16(false, afrag1, false, bfrag,
                                                   (short)0, acc1, false, false);
    WAIT_STAGE();        // next tile resident in the other buffer
    __syncthreads();     // all waves done reading `cur` before it is re-staged
    cur ^= 1;
  }

  const int ncol = n0 + nsub * 16 + l15;
  const float bv = bias ? bias[ncol] : 0.0f;
#pragma unroll
  for (int r = 0; r < 8; ++r) {
    int mr0 = m0 + msub0 * 16 + r + lh * 8;
    int mr1 = mr0 + 16;
    float v0 = acc0[r] + bv, v1 = acc1[r] + bv;
    if (act) { v0 = fmaxf(v0, 0.0f); v1 = fmaxf(v1, 0.0f); }
    C[(size_t)mr0 * N + ncol] = v0;
    C[(size_t)mr1 * N + ncol] = v1;
    if (Cbf) {
      Cbf[(size_t)mr0 * N + ncol] = (__bf16)v0;
      Cbf[(size_t)mr1 * N + ncol] = (__bf16)v1;
    }
  }
}

// ---------------------------------------------------------------------------
// Attention: one wave32 per edge; alpha[h] = SCALE * q[dst]·(k[src]+e)
// amax via monotone-int atomicMax.
// ---------------------------------------------------------------------------
__device__ __forceinline__ int fkey(float f) {
  int b = __float_as_int(f);
  return b >= 0 ? b : (b ^ 0x7fffffff);
}
__device__ __forceinline__ float funkey(int k) {
  int b = k >= 0 ? k : (k ^ 0x7fffffff);
  return __int_as_float(b);
}

__global__ __launch_bounds__(256) void k_alpha_max(
    const float* __restrict__ QKVS, const float* __restrict__ Ef,
    const int* __restrict__ edge_index, float* __restrict__ alpha,
    int* __restrict__ amaxbits) {
  int gw = (blockIdx.x * 256 + threadIdx.x) >> 5;
  int lane = threadIdx.x & 31;
  if (gw >= BB * EE) return;
  int b = gw / EE, e = gw - b * EE;
  int src = edge_index[(size_t)b * 2 * EE + e];
  int dst = edge_index[(size_t)b * 2 * EE + EE + e];
  const float* qrow = QKVS + ((size_t)b * NN + dst) * QK;
  const float* krow = QKVS + ((size_t)b * NN + src) * QK + 1024;
  const float* erow = Ef + ((size_t)b * EE + e) * HD4;
#pragma unroll
  for (int h = 0; h < 4; ++h) {
    int base = h * 256 + lane * 8;
    float s = 0.0f;
#pragma unroll
    for (int d = 0; d < 8; ++d) s += qrow[base + d] * (krow[base + d] + erow[base + d]);
#pragma unroll
    for (int off = 16; off; off >>= 1) s += __shfl_xor(s, off, 32);
    if (lane == 0) {
      s *= SCALE_;
      alpha[((size_t)b * EE + e) * 4 + h] = s;
      atomicMax(&amaxbits[((size_t)b * NN + dst) * 4 + h], fkey(s));
    }
  }
}

__global__ void k_ex_denom(const float* __restrict__ alpha,
                           const int* __restrict__ edge_index,
                           const int* __restrict__ amaxbits,
                           float* __restrict__ exbuf, float* __restrict__ denom) {
  int idx = blockIdx.x * 256 + threadIdx.x;           // (b,e,h)
  if (idx >= BB * EE * 4) return;
  int h = idx & 3, be = idx >> 2;
  int b = be / EE, e = be - b * EE;
  int dst = edge_index[(size_t)b * 2 * EE + EE + e];
  float amax = funkey(amaxbits[((size_t)b * NN + dst) * 4 + h]);
  float ex = __expf(alpha[idx] - amax);
  exbuf[idx] = ex;
  atomicAdd(&denom[((size_t)b * NN + dst) * 4 + h], ex);
}

__global__ __launch_bounds__(256) void k_aggregate(
    const float* __restrict__ QKVS, const float* __restrict__ Ef,
    const int* __restrict__ edge_index, const float* __restrict__ exbuf,
    const float* __restrict__ denom, float* __restrict__ agg) {
  int gw = (blockIdx.x * 256 + threadIdx.x) >> 5;
  int lane = threadIdx.x & 31;
  if (gw >= BB * EE) return;
  int b = gw / EE, e = gw - b * EE;
  int src = edge_index[(size_t)b * 2 * EE + e];
  int dst = edge_index[(size_t)b * 2 * EE + EE + e];
  const float* vrow = QKVS + ((size_t)b * NN + src) * QK + 2048;
  const float* erow = Ef + ((size_t)b * EE + e) * HD4;
  float* arow = agg + ((size_t)b * NN + dst) * HD4;
#pragma unroll
  for (int h = 0; h < 4; ++h) {
    float w = exbuf[((size_t)b * EE + e) * 4 + h] /
              denom[((size_t)b * NN + dst) * 4 + h];
    int base = h * 256 + lane * 8;
#pragma unroll
    for (int d = 0; d < 8; ++d)
      atomicAdd(&arow[base + d], (vrow[base + d] + erow[base + d]) * w);
  }
}

// h = relu(agg + skip) -> bf16 (feeds W1 WMMA GEMM)
__global__ void k_h_relu(const float* __restrict__ agg, const float* __restrict__ QKVS,
                         __bf16* __restrict__ hbf) {
  int idx = blockIdx.x * 256 + threadIdx.x;
  if (idx >= BB * NN * HD4) return;
  int row = idx >> 10, col = idx & 1023;
  float v = agg[idx] + QKVS[(size_t)row * QK + 3072 + col];
  hbf[idx] = (__bf16)fmaxf(v, 0.0f);
}

// ---------------------------------------------------------------------------
// GRU: single persistent workgroup. h[16,256] in LDS (f32 + bf16 mirror);
// each of 8 waves owns o-columns [32w,32w+32) and its r/z/n gate tiles, so the
// recurrent [16,256]x[256,768] matmul per step is 48 WMMAs/wave (6 tiles x 8 K-steps) with no
// cross-wave exchange for the gate nonlinearity.
// ---------------------------------------------------------------------------
__global__ __launch_bounds__(256) void k_gru(
    const float* __restrict__ gx,        // [B*L, 768] (b_ih already added)
    const __bf16* __restrict__ Whh,      // [768, 256] row-major == [N][K]
    const float* __restrict__ bhh, float* __restrict__ hout) {
  __shared__ float  hF[16 * 256];
  __shared__ __bf16 hB[16 * 256];
  const int tid = threadIdx.x, wave = tid >> 5, lane = tid & 31;
  const int lh = (lane >> 4) & 1, l15 = lane & 15;
  for (int i = tid; i < 16 * 256; i += 256) { hF[i] = 0.0f; hB[i] = (__bf16)0.0f; }
  __syncthreads();

  for (int t = 0; t < LL; ++t) {
    v8f acc[6];
#pragma unroll
    for (int i = 0; i < 6; ++i) { v8f z = {}; acc[i] = z; }

    for (int k0 = 0; k0 < 256; k0 += 32) {
      v16bf af;
#pragma unroll
      for (int i = 0; i < 16; ++i) {
        int j = i >> 1, h = i & 1;
        int kk = k0 + lh * 16 + i; (void)j; (void)h;
        af[i] = hB[l15 * 256 + kk];          // A row = batch index
      }
#pragma unroll
      for (int g = 0; g < 3; ++g) {
#pragma unroll
        for (int ot = 0; ot < 2; ++ot) {
          int ncol = (g * 16 + wave * 2 + ot) * 16 + l15;   // gate g, o-tile
          v16bf bf = *(const v16bf*)&Whh[(size_t)ncol * 256 + k0 + lh * 16];
          acc[g * 2 + ot] = __builtin_amdgcn_wmma_f32_16x16x32_bf16(
              false, af, false, bf, (short)0, acc[g * 2 + ot], false, false);
        }
      }
    }

    float hnew[2][8];
#pragma unroll
    for (int ot = 0; ot < 2; ++ot) {
      int o = wave * 32 + ot * 16 + l15;
#pragma unroll
      for (int rr = 0; rr < 8; ++rr) {
        int m = rr + lh * 8;                          // batch row
        const float* gxr = gx + ((size_t)(m * LL + t)) * 768;
        float hr = acc[0 + ot][rr] + bhh[o];
        float hz = acc[2 + ot][rr] + bhh[256 + o];
        float hn = acc[4 + ot][rr] + bhh[512 + o];
        float r_ = sigm(gxr[o] + hr);
        float z_ = sigm(gxr[256 + o] + hz);
        float c  = tanhf(gxr[512 + o] + r_ * hn);
        float hp = hF[m * 256 + o];
        hnew[ot][rr] = (1.0f - z_) * c + z_ * hp;
      }
    }
    __syncthreads();   // all reads of hB/hF done
#pragma unroll
    for (int ot = 0; ot < 2; ++ot) {
      int o = wave * 32 + ot * 16 + l15;
#pragma unroll
      for (int rr = 0; rr < 8; ++rr) {
        int m = rr + lh * 8;
        hF[m * 256 + o] = hnew[ot][rr];
        hB[m * 256 + o] = (__bf16)hnew[ot][rr];
      }
    }
    __syncthreads();
  }
  for (int i = tid; i < 16 * 256; i += 256) hout[i] = hF[i];
}

// ---------------------------------------------------------------------------
// Final scores: wave per (b,a) row
// ---------------------------------------------------------------------------
__global__ __launch_bounds__(256) void k_scores(
    const float* __restrict__ proj, const float* __restrict__ inv,
    const float* __restrict__ aaF, const float* __restrict__ Wab,
    const float* __restrict__ bab, const float* __restrict__ mask,
    float* __restrict__ out) {
  int gw = (blockIdx.x * 256 + threadIdx.x) >> 5;
  int lane = threadIdx.x & 31;
  if (gw >= BB * AAC) return;
  int b = gw / AAC;
  float s = 0.0f;
#pragma unroll
  for (int i = 0; i < 8; ++i) {
    int d = lane * 8 + i;
    s += proj[(size_t)gw * DD + d] * inv[b * DD + d];
    s += aaF[(size_t)gw * DD + d] * Wab[d];
  }
#pragma unroll
  for (int off = 16; off; off >>= 1) s += __shfl_xor(s, off, 32);
  if (lane == 0)
    out[gw] = s + bab[0] + fmaxf(__logf(mask[gw]), -3.402823466e38f);
}

// ---------------------------------------------------------------------------
extern "C" void kernel_launch(void* const* d_in, const int* in_sizes, int n_in,
                              void* d_out, int out_size, void* d_ws, size_t ws_size,
                              hipStream_t stream) {
  const int*   node_tokens = (const int*)d_in[0];
  const int*   edge_tokens = (const int*)d_in[1];
  const int*   edge_index  = (const int*)d_in[2];
  const int*   inv_token   = (const int*)d_in[3];
  const int*   inv_node    = (const int*)d_in[4];
  const int*   aa_token    = (const int*)d_in[5];
  const int*   aa_node     = (const int*)d_in[6];
  const float* action_mask = (const float*)d_in[7];
  const float* emb   = (const float*)d_in[8];
  const float* Wq = (const float*)d_in[9];   const float* bq = (const float*)d_in[10];
  const float* Wk = (const float*)d_in[11];  const float* bk = (const float*)d_in[12];
  const float* Wv = (const float*)d_in[13];  const float* bv = (const float*)d_in[14];
  const float* We = (const float*)d_in[15];
  const float* Wskip = (const float*)d_in[16]; const float* bskip = (const float*)d_in[17];
  const float* W1 = (const float*)d_in[18];  const float* b1 = (const float*)d_in[19];
  const float* Wp = (const float*)d_in[20];  const float* bp = (const float*)d_in[21];
  const float* W_ih = (const float*)d_in[22]; const float* W_hh = (const float*)d_in[23];
  const float* b_ih = (const float*)d_in[24]; const float* b_hh = (const float*)d_in[25];
  const float* Wab = (const float*)d_in[26]; const float* bab = (const float*)d_in[27];
  const float* Wap = (const float*)d_in[28]; const float* bap = (const float*)d_in[29];
  float* out = (float*)d_out;

  char* ws = (char*)d_ws;
  size_t off = 0;
  auto alloc = [&](size_t bytes) -> void* {
    off = (off + 255) & ~(size_t)255;
    void* p = ws + off; off += bytes; return p;
  };

  __bf16* Wcat   = (__bf16*)alloc((size_t)4096 * 256 * 2);
  float*  bcat   = (float*) alloc(4096 * 4);
  __bf16* We_t   = (__bf16*)alloc((size_t)1024 * 256 * 2);
  __bf16* W1_t   = (__bf16*)alloc((size_t)256 * 1024 * 2);
  __bf16* Wp_t   = (__bf16*)alloc(256 * 256 * 2);
  __bf16* Wap_t  = (__bf16*)alloc(256 * 256 * 2);
  __bf16* Wih_b  = (__bf16*)alloc((size_t)768 * 256 * 2);
  __bf16* Whh_b  = (__bf16*)alloc((size_t)768 * 256 * 2);
  __bf16* x_bf   = (__bf16*)alloc((size_t)BB * NN * DD * 2);
  __bf16* ea_bf  = (__bf16*)alloc((size_t)BB * EE * DD * 2);
  float*  QKVS   = (float*) alloc((size_t)BB * NN * QK * 4);
  float*  Ef     = (float*) alloc((size_t)BB * EE * HD4 * 4);
  float*  alpha  = (float*) alloc((size_t)BB * EE * 4 * 4);
  int*    amaxb  = (int*)   alloc((size_t)BB * NN * 4 * 4);
  float*  denom  = (float*) alloc((size_t)BB * NN * 4 * 4);
  float*  exbuf  = (float*) alloc((size_t)BB * EE * 4 * 4);
  float*  agg    = (float*) alloc((size_t)BB * NN * HD4 * 4);
  __bf16* h_bf   = (__bf16*)alloc((size_t)BB * NN * HD4 * 2);
  float*  graph  = (float*) alloc((size_t)BB * NN * DD * 4);
  __bf16* imx_bf = (__bf16*)alloc((size_t)BB * LL * DD * 2);
  float*  invh_f = (float*) alloc((size_t)BB * LL * DD * 4);
  __bf16* invh_b = (__bf16*)alloc((size_t)BB * LL * DD * 2);
  float*  gx     = (float*) alloc((size_t)BB * LL * 768 * 4);
  float*  hout   = (float*) alloc(BB * DD * 4);
  __bf16* amx_bf = (__bf16*)alloc((size_t)BB * AAC * DD * 2);
  float*  amx_f  = (float*) alloc((size_t)BB * AAC * DD * 4);
  float*  proj   = (float*) alloc((size_t)BB * AAC * DD * 4);

  // ---- weight prep -------------------------------------------------------
  k_convert_t<<<1024, 256, 0, stream>>>(Wq,    Wcat + (size_t)0 * 256,    256, 1024);
  k_convert_t<<<1024, 256, 0, stream>>>(Wk,    Wcat + (size_t)1024 * 256, 256, 1024);
  k_convert_t<<<1024, 256, 0, stream>>>(Wv,    Wcat + (size_t)2048 * 256, 256, 1024);
  k_convert_t<<<1024, 256, 0, stream>>>(Wskip, Wcat + (size_t)3072 * 256, 256, 1024);
  (void)hipMemcpyAsync(bcat + 0,    bq,    1024 * 4, hipMemcpyDeviceToDevice, stream);
  (void)hipMemcpyAsync(bcat + 1024, bk,    1024 * 4, hipMemcpyDeviceToDevice, stream);
  (void)hipMemcpyAsync(bcat + 2048, bv,    1024 * 4, hipMemcpyDeviceToDevice, stream);
  (void)hipMemcpyAsync(bcat + 3072, bskip, 1024 * 4, hipMemcpyDeviceToDevice, stream);
  k_convert_t<<<1024, 256, 0, stream>>>(We, We_t, 256, 1024);
  k_convert_t<<<1024, 256, 0, stream>>>(W1, W1_t, 1024, 256);
  k_convert_t<<<256,  256, 0, stream>>>(Wp, Wp_t, 256, 256);
  k_convert_t<<<256,  256, 0, stream>>>(Wap, Wap_t, 256, 256);
  k_convert<<<768, 256, 0, stream>>>(W_ih, Wih_b, 768 * 256);
  k_convert<<<768, 256, 0, stream>>>(W_hh, Whh_b, 768 * 256);

  // ---- embeddings --------------------------------------------------------
  k_gather<<<16000, 256, 0, stream>>>(node_tokens, emb, x_bf, BB * NN);
  k_gather<<<64000, 256, 0, stream>>>(edge_tokens, emb, ea_bf, BB * EE);

  // ---- big WMMA GEMMs ----------------------------------------------------
  k_gemm<<<dim3(64, 250), 256, 0, stream>>>(x_bf, Wcat, bcat, QKVS, nullptr,
                                            BB * NN, 4096, 256, 0);
  k_gemm<<<dim3(16, 1000), 256, 0, stream>>>(ea_bf, We_t, nullptr, Ef, nullptr,
                                             BB * EE, 1024, 256, 0);

  // ---- attention ---------------------------------------------------------
  k_fill_int<<<250, 256, 0, stream>>>(amaxb, (int)0x80000000, BB * NN * 4);
  (void)hipMemsetAsync(denom, 0, (size_t)BB * NN * 4 * 4, stream);
  (void)hipMemsetAsync(agg,   0, (size_t)BB * NN * HD4 * 4, stream);
  k_alpha_max<<<8000, 256, 0, stream>>>(QKVS, Ef, edge_index, alpha, amaxb);
  k_ex_denom<<<1000, 256, 0, stream>>>(alpha, edge_index, amaxb, exbuf, denom);
  k_aggregate<<<8000, 256, 0, stream>>>(QKVS, Ef, edge_index, exbuf, denom, agg);
  k_h_relu<<<64000, 256, 0, stream>>>(agg, QKVS, h_bf);

  // ---- graph_repr --------------------------------------------------------
  k_gemm<<<dim3(4, 250), 256, 0, stream>>>(h_bf, W1_t, b1, graph, nullptr,
                                           BB * NN, 256, 1024, 1);

  // ---- invariant path + GRU ---------------------------------------------
  k_mixed<<<1024, 256, 0, stream>>>(inv_token, inv_node, emb, graph,
                                    imx_bf, nullptr, BB * LL, LL);
  k_gemm<<<dim3(4, 16), 256, 0, stream>>>(imx_bf, Wp_t, bp, invh_f, invh_b,
                                          BB * LL, 256, 256, 1);
  k_gemm<<<dim3(12, 16), 256, 0, stream>>>(invh_b, Wih_b, b_ih, gx, nullptr,
                                           BB * LL, 768, 256, 0);
  k_gru<<<1, 256, 0, stream>>>(gx, Whh_b, b_hh, hout);

  // ---- action scoring ----------------------------------------------------
  k_mixed<<<2048, 256, 0, stream>>>(aa_token, aa_node, emb, graph,
                                    amx_bf, amx_f, BB * AAC, AAC);
  k_gemm<<<dim3(4, 32), 256, 0, stream>>>(amx_bf, Wap_t, bap, proj, nullptr,
                                          BB * AAC, 256, 256, 0);
  k_scores<<<256, 256, 0, stream>>>(proj, hout, amx_f, Wab, bab, action_mask, out);
  (void)in_sizes; (void)n_in; (void)out_size; (void)ws_size;
}
